// BatchMultiHeadGraphAttention_34359738433
// MI455X (gfx1250) — matrix operand order repaired and checked by reading, not compiled, not run on hardware
//
#include <hip/hip_runtime.h>
#include <hip/hip_bf16.h>

typedef __attribute__((ext_vector_type(16))) _Float16 v16h;
typedef __attribute__((ext_vector_type(8)))  float    v8f;

#define N_HEAD 4
#define F_IN   32
#define F_OUT  32
#define DEG    16

// ---------------------------------------------------------------------------
// K0: pre-swizzle w (4,32,32) f32 into the f16 B-fragment lane layout.
// frag = (hd*2 + t), lane in [0,32), element e in [0,16):
//   B[K][Ncol] with K = (e&8)*2 + (lane>>4)*8 + (e&7), Ncol = t*16 + (lane&15).
// ---------------------------------------------------------------------------
__global__ void gat_prep_w(const float* __restrict__ w, _Float16* __restrict__ wsw)
{
    const int idx   = threadIdx.x;          // 0..255
    const int lane  = idx & 31;
    const int t     = (idx >> 5) & 1;
    const int hd    = idx >> 6;
    const int rlane = lane & 15;
    const int hi    = lane >> 4;
    const float* wh = w + (size_t)hd * F_IN * F_OUT;
    _Float16* dstp  = wsw + (size_t)idx * 16;
#pragma unroll
    for (int e = 0; e < 16; ++e)
        dstp[e] = (_Float16)wh[(size_t)((e & 8) * 2 + hi * 8 + (e & 7)) * F_OUT + t * 16 + rlane];
}

// ---------------------------------------------------------------------------
// K1: hp[h][n][o] = sum_f h[n][f] * w[h][f][o]  via v_wmma_f32_16x16x32_f16.
// One wave per block, one full 16-node tile per wave: tile = blockIdx.x.
// NO bounds checks, NO predication -> EXEC all-ones, clean store clauses.
// ---------------------------------------------------------------------------
__global__ void __launch_bounds__(32)
gat_proj_wmma(const float* __restrict__ h,
              const _Float16* __restrict__ wsw,
              float* __restrict__ hp, int n)
{
    const int lane  = threadIdx.x;          // 0..31
    const int row0  = blockIdx.x * 16;      // always a full tile
    const int rlane = lane & 15;
    const int hi    = lane >> 4;
    const int row   = row0 + rlane;

    // A fragment: lanes 0-15 rows M=0-15; elems 0..7 -> K=hi*8+e,
    // elems 8..15 -> K=16+hi*8+(e-8).  Two 32B contiguous segments -> b128 x4.
    const float4* hrow4 = (const float4*)(h + (size_t)row * F_IN);
    float4 p0 = hrow4[hi * 2 + 0];
    float4 p1 = hrow4[hi * 2 + 1];
    float4 p2 = hrow4[4 + hi * 2 + 0];
    float4 p3 = hrow4[4 + hi * 2 + 1];
    v16h a;
    a[0]  = (_Float16)p0.x; a[1]  = (_Float16)p0.y; a[2]  = (_Float16)p0.z; a[3]  = (_Float16)p0.w;
    a[4]  = (_Float16)p1.x; a[5]  = (_Float16)p1.y; a[6]  = (_Float16)p1.z; a[7]  = (_Float16)p1.w;
    a[8]  = (_Float16)p2.x; a[9]  = (_Float16)p2.y; a[10] = (_Float16)p2.z; a[11] = (_Float16)p2.w;
    a[12] = (_Float16)p3.x; a[13] = (_Float16)p3.y; a[14] = (_Float16)p3.z; a[15] = (_Float16)p3.w;

    const v16h* bfrag = (const v16h*)wsw;   // [(hd*2+t)*32 + lane]

#pragma unroll
    for (int hd = 0; hd < N_HEAD; ++hd) {
        float* hph = hp + (size_t)hd * n * F_OUT;
#pragma unroll
        for (int t = 0; t < 2; ++t) {
            v16h b = bfrag[(size_t)(hd * 2 + t) * 32 + lane];   // 32B coalesced

            v8f c = {};
            c = __builtin_amdgcn_wmma_f32_16x16x32_f16(
                    /*neg_a=*/false, a, /*neg_b=*/false, b,
                    /*c_mod=*/(short)0, c, /*reuse_a=*/false, /*reuse_b=*/false);

            // D layout: VGPR j -> M=j (lanes 0-15) / M=j+8 (lanes 16-31); col=rlane.
#pragma unroll
            for (int j = 0; j < 8; ++j)
                hph[(size_t)(row0 + j + 8 * hi) * F_OUT + t * 16 + rlane] = c[j];
        }
    }
}

// ---------------------------------------------------------------------------
// K1b: scalar tail for the last n%16 rows (not launched when n%16==0).
// thread -> (hd, row, col); fp32 dot over F_IN.
// ---------------------------------------------------------------------------
__global__ void gat_proj_tail(const float* __restrict__ h,
                              const float* __restrict__ w,
                              float* __restrict__ hp, int n, int row_start)
{
    int idx  = blockIdx.x * blockDim.x + threadIdx.x;
    int rem  = n - row_start;
    int tot  = rem * N_HEAD * F_OUT;
    if (idx >= tot) return;
    int col  = idx & (F_OUT - 1);
    int r    = (idx / F_OUT) % rem + row_start;
    int hd   = idx / (F_OUT * rem);
    const float* hrow = h + (size_t)r * F_IN;
    const float* wh   = w + (size_t)hd * F_IN * F_OUT;
    float acc = 0.f;
#pragma unroll
    for (int f = 0; f < F_IN; ++f)
        acc = fmaf(hrow[f], wh[(size_t)f * F_OUT + col], acc);
    hp[((size_t)hd * n + r) * F_OUT + col] = acc;
}

// ---------------------------------------------------------------------------
// K2: a_src[h*n+i] = hp[h][i][:] . fc_w[0:32],  a_dst = . fc_w[32:64]
// ---------------------------------------------------------------------------
__global__ void gat_attn_coeff(const float* __restrict__ hp,
                               const float* __restrict__ fc_w,
                               float* __restrict__ a_src,
                               float* __restrict__ a_dst, int n)
{
    int idx = blockIdx.x * blockDim.x + threadIdx.x;   // h*n + node
    if (idx >= N_HEAD * n) return;
    const float4* row4 = (const float4*)(hp + (size_t)idx * F_OUT);
    float s = 0.f, d = 0.f;
#pragma unroll
    for (int q = 0; q < 8; ++q) {
        float4 v = row4[q];
        s = fmaf(v.x, fc_w[4*q+0], s); s = fmaf(v.y, fc_w[4*q+1], s);
        s = fmaf(v.z, fc_w[4*q+2], s); s = fmaf(v.w, fc_w[4*q+3], s);
        d = fmaf(v.x, fc_w[F_OUT+4*q+0], d); d = fmaf(v.y, fc_w[F_OUT+4*q+1], d);
        d = fmaf(v.z, fc_w[F_OUT+4*q+2], d); d = fmaf(v.w, fc_w[F_OUT+4*q+3], d);
    }
    a_src[idx] = s;
    a_dst[idx] = d;
}

// ---------------------------------------------------------------------------
// K3: one wave per node. src = repeat(arange(n), 16) => edges 16i..16i+15
// belong to node i. 16-wide softmax via lane shuffles, then coalesced
// gather-FMA of hp rows (L2-resident; 192MB L2 >> 25.6MB hp).
// ---------------------------------------------------------------------------
__global__ void gat_softmax_agg(const float* __restrict__ hp,
                                const float* __restrict__ a_src,
                                const float* __restrict__ a_dst,
                                const int*   __restrict__ dst,
                                const float* __restrict__ fc_b,
                                const float* __restrict__ bias,
                                float* __restrict__ out, int n)
{
    const int lane = threadIdx.x & 31;
    const int node = blockIdx.x * (blockDim.x >> 5) + (threadIdx.x >> 5);
    if (node >= n) return;

    const int   j  = lane & 15;                 // edge slot, replicated both halves
    const int   d  = dst[(size_t)node * DEG + j];
    const float b0 = fc_b[0];

    float acc = 0.f;
    for (int hd = 0; hd < N_HEAD; ++hd) {
        const float* __restrict__ hph = hp + (size_t)hd * n * F_OUT;
        __builtin_prefetch(hph + (size_t)d * F_OUT, 0, 0);   // global_prefetch_b8

        float e = a_src[(size_t)hd * n + node] + a_dst[(size_t)hd * n + d] + b0;
        e = e > 0.f ? e : 0.2f * e;             // leaky_relu(0.2)

        float m = e;
#pragma unroll
        for (int x = 1; x <= 8; x <<= 1) m = fmaxf(m, __shfl_xor(m, x, 32));
        float ex = __expf(e - m);
        float s  = ex;
#pragma unroll
        for (int x = 1; x <= 8; x <<= 1) s += __shfl_xor(s, x, 32);
        float att = ex / s;

#pragma unroll
        for (int jj = 0; jj < DEG; ++jj) {
            float aj = __shfl(att, jj, 32);
            int   dj = __shfl(d,   jj, 32);
            acc = fmaf(aj, hph[(size_t)dj * F_OUT + lane], acc);
        }
    }
    out[(size_t)node * F_OUT + lane] = acc * (1.f / N_HEAD) + bias[lane];
}

// ---------------------------------------------------------------------------
extern "C" void kernel_launch(void* const* d_in, const int* in_sizes, int n_in,
                              void* d_out, int out_size, void* d_ws, size_t ws_size,
                              hipStream_t stream)
{
    const float* h    = (const float*)d_in[0];          // (n, 32)
    const int*   ei   = (const int*)  d_in[1];          // (2, n*16): src then dst
    const float* w    = (const float*)d_in[2];          // (4, 32, 32)
    const float* fc_w = (const float*)d_in[3];          // (1, 64)
    const float* fc_b = (const float*)d_in[4];          // (1,)
    const float* bias = (const float*)d_in[5];          // (32,)
    float*       out  = (float*)d_out;

    const int n       = in_sizes[0] / F_IN;             // 50000
    const size_t nedg = (size_t)n * DEG;
    const int* dst    = ei + nedg;                      // edge_index[1]

    // workspace layout (all offsets 32B-aligned)
    char* ws = (char*)d_ws;
    float*    hp    = (float*)ws;                                       // 4*n*32 f32
    float*    a_src = (float*)(ws + (size_t)N_HEAD * n * F_OUT * 4);    // 4*n f32
    float*    a_dst = a_src + (size_t)N_HEAD * n;                       // 4*n f32
    _Float16* wsw   = (_Float16*)(a_dst + (size_t)N_HEAD * n);          // 256*16 f16

    // K0: swizzle w into B-fragment layout (one 256-thread block)
    gat_prep_w<<<1, 256, 0, stream>>>(w, wsw);

    // K1: WMMA projection — one wave per full 16-node tile, branch-free
    const int full_tiles = n / 16;
    if (full_tiles > 0)
        gat_proj_wmma<<<full_tiles, 32, 0, stream>>>(h, wsw, hp, n);

    // K1b: scalar tail for n%16 rows (0 blocks for n=50000 -> not launched)
    const int rem = n - full_tiles * 16;
    if (rem > 0) {
        int tot = rem * N_HEAD * F_OUT;
        gat_proj_tail<<<(tot + 255) / 256, 256, 0, stream>>>(h, w, hp, n,
                                                             full_tiles * 16);
    }

    // K2: attention coefficients
    {
        int total  = N_HEAD * n;
        int blocks = (total + 255) / 256;
        gat_attn_coeff<<<blocks, 256, 0, stream>>>(hp, fc_w, a_src, a_dst, n);
    }
    // K3: softmax + aggregation — one wave per node, 8 waves/block
    {
        int blocks = (n + 7) / 8;
        gat_softmax_agg<<<blocks, 256, 0, stream>>>(hp, a_src, a_dst, dst,
                                                    fc_b, bias, out, n);
    }
}
